// Seq2SeqTransformer_21543555956815
// MI455X (gfx1250) — compile-verified, matrix operand order checked by reading, not run on hardware
//
#include <hip/hip_runtime.h>
#include <stdint.h>

#define LL 2048
#define NN 4
#define EE 512
#define HH 8
#define DD 64

typedef __attribute__((ext_vector_type(16))) _Float16 v16h;
typedef __attribute__((ext_vector_type(8)))  float    v8f;

union Frag16 { v16h v; _Float16 h[16]; uint32_t u[8]; };

// A-fragment (16x32 f16) from row-major f16 tile.  lane&15 = row (M),
// lane>>4 selects K half; VGPR j holds K = (j>>2)*16 + half*8 + (j&3)*2 (+1).
// B-fragments (32x16) load with the SAME pattern from the row-major transpose
// of B (rows = output columns), per ISA 7.12.2 symmetry.
__device__ __forceinline__ v16h load_frag_f16(const _Float16* base, int stride, int lane) {
  Frag16 f;
  const int m  = lane & 15;
  const int hh = (lane >> 4) & 1;
  const _Float16* row = base + m * stride;
#pragma unroll
  for (int j = 0; j < 8; ++j) {
    int k = ((j >> 2) << 4) + (hh << 3) + ((j & 3) << 1);
    f.u[j] = *reinterpret_cast<const uint32_t*>(row + k);
  }
  return f.v;
}

__device__ __forceinline__ v16h load_frag_f32cvt(const float* base, int stride, int lane) {
  Frag16 f;
  const int m  = lane & 15;
  const int hh = (lane >> 4) & 1;
  const float* row = base + m * stride;
#pragma unroll
  for (int j = 0; j < 8; ++j) {
    int k = ((j >> 2) << 4) + (hh << 3) + ((j & 3) << 1);
    f.h[2 * j]     = (_Float16)row[k];
    f.h[2 * j + 1] = (_Float16)row[k + 1];
  }
  return f.v;
}

__device__ __forceinline__ v8f wmma16(v16h a, v16h b, v8f c) {
  return __builtin_amdgcn_wmma_f32_16x16x32_f16(false, a, false, b, (short)0, c, false, false);
}

__device__ __forceinline__ float rowmax16(float v) {
#pragma unroll
  for (int off = 1; off < 16; off <<= 1) v = fmaxf(v, __shfl_xor(v, off, 32));
  return v;
}
__device__ __forceinline__ float rowsum16(float v) {
#pragma unroll
  for (int off = 1; off < 16; off <<= 1) v += __shfl_xor(v, off, 32);
  return v;
}

__device__ __forceinline__ uint4 ldg4(const _Float16* p) {
  return *reinterpret_cast<const uint4*>(p);
}
__device__ __forceinline__ void sts4(_Float16* p, uint4 v) {
  *reinterpret_cast<uint4*>(p) = v;
}

// ---------------- 1) weight f32 -> f16 conversion ----------------
__global__ void cvt_weights_kernel(const float* __restrict__ W, const float* __restrict__ WO,
                                   _Float16* __restrict__ Wh, _Float16* __restrict__ WOh) {
  const int nW  = 3 * EE * EE;
  const int nWO = EE * EE;
  for (int i = blockIdx.x * blockDim.x + threadIdx.x; i < nW + nWO;
       i += gridDim.x * blockDim.x) {
    if (i < nW) Wh[i] = (_Float16)W[i];
    else        WOh[i - nW] = (_Float16)WO[i - nW];
  }
}

// ---------------- 2) fused QKV projection (WMMA, LDS-staged weights) ------
// Block = 128 rows x 64 cols; 8 waves each own 16 rows and share the W tile,
// which is double-buffered in LDS (4KB/stage) with register prefetch.
__global__ void __launch_bounds__(256) proj_kernel(
    const float* __restrict__ q_in, const float* __restrict__ k_in,
    const float* __restrict__ v_in, const float* __restrict__ bias,
    const _Float16* __restrict__ Wh, const float* __restrict__ p_par,
    const float* __restrict__ shift, _Float16* __restrict__ QS,
    _Float16* __restrict__ KK, _Float16* __restrict__ VPT) {
  __shared__ __align__(16) _Float16 Wt[2][64 * 32];
  const int tid  = threadIdx.x;
  const int lane = tid & 31;
  const int wave = tid >> 5;
  const int rowBlk   = blockIdx.x / 24;       // 64 blocks of 128 rows
  const int colStrip = blockIdx.x % 24;       // 24 strips of 64 cols
  const int row0 = rowBlk * 128 + wave * 16;
  const int col0 = colStrip * 64;
  const float* X = (col0 < EE) ? q_in : (col0 < 2 * EE ? k_in : v_in);

  const int wrow = tid >> 2, wch = tid & 3;   // staging: 64 rows x 4 chunks of 8 halves
  const _Float16* wsrc = Wh + (size_t)(col0 + wrow) * EE + wch * 8;

  uint4 wreg = ldg4(wsrc);
  sts4(&Wt[0][tid * 8], wreg);
  __syncthreads();

  v8f acc[4] = {};
  for (int ks = 0; ks < 16; ++ks) {
    const int cur = ks & 1;
    const int kk = ks * 32;
    if (ks + 1 < 16) wreg = ldg4(wsrc + kk + 32);          // prefetch next stage
    v16h a = load_frag_f32cvt(X + (size_t)row0 * EE + kk, EE, lane);
#pragma unroll
    for (int t = 0; t < 4; ++t) {
      v16h b = load_frag_f16(&Wt[cur][t * 16 * 32], 32, lane);
      acc[t] = wmma16(a, b, acc[t]);
    }
    if (ks + 1 < 16) sts4(&Wt[cur ^ 1][tid * 8], wreg);
    __syncthreads();
  }

  const int half = lane >> 4;
  const int nc = lane & 15;
#pragma unroll
  for (int t = 0; t < 4; ++t) {
    const int eo = col0 + t * 16 + nc;
    const float bv = bias[eo];
#pragma unroll
    for (int vr = 0; vr < 8; ++vr) {
      const int r = row0 + vr + 8 * half;
      const int l = r >> 2;          // row = l*N + n, N = 4
      const int n = r & 3;
      const float y = acc[t][vr] + bv;
      if (eo < EE) {
        const int e = eo, h = e >> 6, d = e & 63;
        QS[(((size_t)(n * HH + h) * LL + l) << 6) + d] = (_Float16)(y * 0.125f);
      } else if (eo < 2 * EE) {
        const int e = eo - EE, h = e >> 6, d = e & 63;
        KK[(((size_t)(n * HH + h) * LL + l) << 6) + d] = (_Float16)y;
      } else {
        const int e = eo - 2 * EE, h = e >> 6, d = e & 63;
        const float vp = __powf(fmaxf(y + shift[e], 1e-6f), p_par[e]);
        VPT[((size_t)(n * HH + h) * DD + d) * LL + l] = (_Float16)vp;
      }
    }
  }
}

// ---------------- 3) flash attention + GeM pooling --------------------
// Block = one (n,h), 128 query rows (8 waves x 16).  K and V^p tiles for the
// current 32-wide S-block are staged ONCE per block in double-buffered LDS
// (2 x 8KB) with register prefetch -> 8x less L2 traffic than per-wave loads.
__global__ void __launch_bounds__(256) attn_kernel(
    const _Float16* __restrict__ QS, const _Float16* __restrict__ KK,
    const _Float16* __restrict__ VPT, const float* __restrict__ p_par,
    const float* __restrict__ shift, _Float16* __restrict__ OH) {
  __shared__ __align__(16) _Float16 Kt[2][32 * 64];
  __shared__ __align__(16) _Float16 Vt[2][64 * 32];
  __shared__ _Float16 pbuf[8][16 * 32];
  const int tid  = threadIdx.x;
  const int lane = tid & 31;
  const int wave = tid >> 5;
  const int nh   = blockIdx.x >> 4;           // 32 (n,h) pairs
  const int qblk = blockIdx.x & 15;           // 16 blocks of 128 query rows
  const int n = nh >> 3, h = nh & 7;
  const int l0 = qblk * 128 + wave * 16;

  const _Float16* Qb = QS + ((size_t)nh * LL + l0) * DD;
  const _Float16* Kb = KK + (size_t)nh * LL * DD;
  const _Float16* Vb = VPT + (size_t)nh * DD * LL;

  const v16h qa0 = load_frag_f16(Qb, DD, lane);       // d = 0..31
  const v16h qa1 = load_frag_f16(Qb + 32, DD, lane);  // d = 32..63

  v8f acc[4] = {};
  float m_run[8], l_run[8];
#pragma unroll
  for (int i = 0; i < 8; ++i) { m_run[i] = -1e30f; l_run[i] = 0.f; }

  _Float16* myp = pbuf[wave];
  const int row_lds = 8 * (lane >> 4);
  const int col_lds = lane & 15;

  // staging: K block 32x64 halves linear; V block 64 rows x 32 halves
  const int vrow = tid >> 2, vch = tid & 3;
  uint4 kreg = ldg4(Kb + tid * 8);
  uint4 vreg = ldg4(Vb + (size_t)vrow * LL + vch * 8);
  sts4(&Kt[0][tid * 8], kreg);
  sts4(&Vt[0][tid * 8], vreg);
  __syncthreads();

  for (int it = 0; it < LL / 32; ++it) {
    const int cur = it & 1;
    const int s0n = (it + 1) * 32;
    if (it + 1 < LL / 32) {                                    // prefetch next
      kreg = ldg4(Kb + (size_t)s0n * DD + tid * 8);
      vreg = ldg4(Vb + (size_t)vrow * LL + s0n + vch * 8);
    }

    // scores tile 16x32 from LDS-staged K
    v8f c0 = {}, c1 = {};
    {
      v16h b = load_frag_f16(&Kt[cur][0], DD, lane);
      c0 = wmma16(qa0, b, c0);
      b = load_frag_f16(&Kt[cur][32], DD, lane);
      c0 = wmma16(qa1, b, c0);
      b = load_frag_f16(&Kt[cur][16 * DD], DD, lane);
      c1 = wmma16(qa0, b, c1);
      b = load_frag_f16(&Kt[cur][16 * DD + 32], DD, lane);
      c1 = wmma16(qa1, b, c1);
    }
    // online softmax (row = vr + 8*(lane>=16); 16-lane xor reductions)
#pragma unroll
    for (int vr = 0; vr < 8; ++vr) {
      float bm = rowmax16(fmaxf(c0[vr], c1[vr]));
      const float nm = fmaxf(m_run[vr], bm);
      const float alpha = __expf(m_run[vr] - nm);
      const float p0 = __expf(c0[vr] - nm);
      const float p1 = __expf(c1[vr] - nm);
      const float rs = rowsum16(p0 + p1);
      l_run[vr] = l_run[vr] * alpha + rs;
      m_run[vr] = nm;
#pragma unroll
      for (int t = 0; t < 4; ++t) acc[t][vr] *= alpha;
      const int row = vr + row_lds;
      myp[row * 32 + col_lds]      = (_Float16)p0;
      myp[row * 32 + col_lds + 16] = (_Float16)p1;
    }
    asm volatile("s_wait_dscnt 0" ::: "memory");   // C->A re-layout via LDS
    const v16h pa = load_frag_f16(myp, 32, lane);
#pragma unroll
    for (int t = 0; t < 4; ++t) {
      v16h vb = load_frag_f16(&Vt[cur][t * 16 * 32], 32, lane);
      acc[t] = wmma16(pa, vb, acc[t]);
    }

    if (it + 1 < LL / 32) {                        // fill other buffer
      sts4(&Kt[cur ^ 1][tid * 8], kreg);
      sts4(&Vt[cur ^ 1][tid * 8], vreg);
    }
    __syncthreads();
  }

  // epilogue: normalize, inverse GeM power, store f16 (L*N, E) row-major
  const int half = lane >> 4;
  const int nc = lane & 15;
  float invl[8];
#pragma unroll
  for (int vr = 0; vr < 8; ++vr) invl[vr] = 1.0f / fmaxf(l_run[vr], 1e-30f);
#pragma unroll
  for (int t = 0; t < 4; ++t) {
    const int d = t * 16 + nc;
    const int e = h * DD + d;
    const float ip = 1.0f / p_par[e];
    const float sh = shift[e];
#pragma unroll
    for (int vr = 0; vr < 8; ++vr) {
      const int l = l0 + vr + 8 * half;
      const float pooled = acc[t][vr] * invl[vr];
      const float val = __powf(fmaxf(pooled, 1e-6f), ip) - sh;
      OH[((size_t)l * NN + n) * EE + e] = (_Float16)val;
    }
  }
}

// ---------------- 4) output projection (WMMA, LDS-staged weights) ---------
__global__ void __launch_bounds__(256) outproj_kernel(
    const _Float16* __restrict__ OH, const _Float16* __restrict__ WOh,
    const float* __restrict__ bias, float* __restrict__ out) {
  __shared__ __align__(16) _Float16 Wt[2][64 * 32];
  const int tid  = threadIdx.x;
  const int lane = tid & 31;
  const int wave = tid >> 5;
  const int rowBlk   = blockIdx.x >> 3;       // 64 blocks of 128 rows
  const int colStrip = blockIdx.x & 7;        // 8 strips of 64 cols
  const int row0 = rowBlk * 128 + wave * 16;
  const int col0 = colStrip * 64;

  const int wrow = tid >> 2, wch = tid & 3;
  const _Float16* wsrc = WOh + (size_t)(col0 + wrow) * EE + wch * 8;

  uint4 wreg = ldg4(wsrc);
  sts4(&Wt[0][tid * 8], wreg);
  __syncthreads();

  v8f acc[4] = {};
  for (int ks = 0; ks < 16; ++ks) {
    const int cur = ks & 1;
    const int kk = ks * 32;
    if (ks + 1 < 16) wreg = ldg4(wsrc + kk + 32);
    v16h a = load_frag_f16(OH + (size_t)row0 * EE + kk, EE, lane);
#pragma unroll
    for (int t = 0; t < 4; ++t) {
      v16h b = load_frag_f16(&Wt[cur][t * 16 * 32], 32, lane);
      acc[t] = wmma16(a, b, acc[t]);
    }
    if (ks + 1 < 16) sts4(&Wt[cur ^ 1][tid * 8], wreg);
    __syncthreads();
  }

  const int half = lane >> 4;
  const int nc = lane & 15;
#pragma unroll
  for (int t = 0; t < 4; ++t) {
    const int c = col0 + t * 16 + nc;
    const float bv = bias[c];
#pragma unroll
    for (int vr = 0; vr < 8; ++vr) {
      const int r = row0 + vr + 8 * half;
      out[(size_t)r * EE + c] = acc[t][vr] + bv;
    }
  }
}

extern "C" void kernel_launch(void* const* d_in, const int* in_sizes, int n_in,
                              void* d_out, int out_size, void* d_ws, size_t ws_size,
                              hipStream_t stream) {
  (void)in_sizes; (void)n_in; (void)out_size; (void)ws_size;
  const float* query = (const float*)d_in[0];
  const float* key   = (const float*)d_in[1];
  const float* value = (const float*)d_in[2];
  const float* ipw   = (const float*)d_in[3];
  const float* ipb   = (const float*)d_in[4];
  const float* opw   = (const float*)d_in[5];
  const float* opb   = (const float*)d_in[6];
  const float* p_par = (const float*)d_in[7];
  const float* shift = (const float*)d_in[8];
  float* out = (float*)d_out;

  const size_t MAT = (size_t)LL * NN * EE;     // 4,194,304 halves
  _Float16* ws  = (_Float16*)d_ws;
  _Float16* Wh  = ws;                          // 3*E*E halves
  _Float16* WOh = Wh + (size_t)3 * EE * EE;    // E*E halves
  _Float16* QS  = WOh + (size_t)EE * EE;       // MAT
  _Float16* KK  = QS + MAT;                    // MAT
  _Float16* VPT = KK + MAT;                    // MAT
  _Float16* OH  = VPT + MAT;                   // MAT  (total ~34 MB)

  cvt_weights_kernel<<<512, 256, 0, stream>>>(ipw, opw, Wh, WOh);
  proj_kernel<<<1536, 256, 0, stream>>>(query, key, value, ipb, Wh, p_par, shift,
                                        QS, KK, VPT);
  attn_kernel<<<512, 256, 0, stream>>>(QS, KK, VPT, p_par, shift, OH);
  outproj_kernel<<<512, 256, 0, stream>>>(OH, WOh, opb, out);
}
